// GDCRewiring_31971736551539
// MI455X (gfx1250) — matrix operand order repaired and checked by reading, not compiled, not run on hardware
//
#include <hip/hip_runtime.h>
#include <hip/hip_bf16.h>

// ---------------------------------------------------------------------------
// GDC rewiring: S = alpha * inv(I - 0.95 * D^-1/2 (A+I) D^-1/2), top-128 per
// column, 64 graphs x 1024 nodes. A is NOT symmetric (directed random edges),
// so we Newton-iterate on W = M^T: X -> 2X - X(WX) converges to inv(M)^T,
// whose rows are the columns of S (contiguous top-k).
// GEMMs: v_wmma_f32_16x16x32_bf16 with 2-term bf16 split (hi*hi+hi*lo+lo*hi).
// Every GEMM operand tile is a contiguous row-slab (epilogues emit transposed
// copies), fed through double-buffered LDS via GLOBAL_LOAD_ASYNC_TO_LDS_B128.
// ---------------------------------------------------------------------------

typedef __attribute__((ext_vector_type(16))) __bf16 v16bf;
typedef __attribute__((ext_vector_type(8)))  float  v8f;
typedef int v4i __attribute__((__vector_size__(4 * sizeof(int))));

#define Gx 64
#define Nx 1024
#define Kx 128
#define BM 128
#define BN 128
#define BK 32
#define LDT 40          // padded LDS row stride (bf16 elems); 80B rows, 16B-aligned chunks
#define NEWTON_ITERS 12

#if defined(__has_builtin)
#if __has_builtin(__builtin_amdgcn_global_load_async_to_lds_b128)
#define HAVE_ASYNC_LDS 1
#endif
#endif
#ifndef HAVE_ASYNC_LDS
#define HAVE_ASYNC_LDS 0
#endif

#define GPTR4 __attribute__((address_space(1))) v4i*
#define SPTR4 __attribute__((address_space(3))) v4i*

__device__ __forceinline__ void wait_async0() {
#if HAVE_ASYNC_LDS
#if __has_builtin(__builtin_amdgcn_s_wait_asynccnt)
  __builtin_amdgcn_s_wait_asynccnt(0);
#else
  asm volatile("s_wait_asynccnt 0x0" ::: "memory");
#endif
#endif
}

__device__ __forceinline__ void split_bf16(float v, __bf16& hi, __bf16& lo) {
  hi = (__bf16)v;
  lo = (__bf16)(v - (float)hi);
}

// ------------------------- setup kernels -----------------------------------

__global__ __launch_bounds__(256) void zero_f32(float* __restrict__ p) {
  size_t e = (size_t)blockIdx.x * 256 + threadIdx.x;
  p[e] = 0.0f;
}

__global__ __launch_bounds__(256) void scatter_edges(const int* __restrict__ ei,
                                                     int Etot,
                                                     float* __restrict__ A) {
  int e = blockIdx.x * 256 + threadIdx.x;
  if (e >= Etot) return;
  int s = ei[e];
  int d = ei[Etot + e];
  int g = s >> 10;
  A[(size_t)g * ((size_t)Nx * Nx) + (size_t)(s - (g << 10)) * Nx + (d - (g << 10))] = 1.0f;
}

__global__ __launch_bounds__(256) void add_diag(float* __restrict__ A) {
  int r = blockIdx.x * 256 + threadIdx.x;            // r = g*N + i
  A[(size_t)r * Nx + (r & (Nx - 1))] += 1.0f;
}

__global__ __launch_bounds__(256) void rowsum_rsqrt(const float* __restrict__ A,
                                                    float* __restrict__ dis) {
  __shared__ float red[256];
  int b = blockIdx.x;                                 // g*N + i
  int t = threadIdx.x;
  const float* row = A + (size_t)b * Nx;
  float s = row[t] + row[t + 256] + row[t + 512] + row[t + 768];
  red[t] = s;
  __syncthreads();
  for (int st = 128; st > 0; st >>= 1) {
    if (t < st) red[t] += red[t + st];
    __syncthreads();
  }
  if (t == 0) dis[b] = rsqrtf(red[0]);
}

// W = M^T:  W[i][j] = delta_ij - 0.95 * dis_j * A[j][i] * dis_i
// (transposed reads hit L2: one graph's A is 4MB << 192MB)
__global__ __launch_bounds__(256) void make_Mt(const float* __restrict__ A,
                                               const float* __restrict__ dis,
                                               __bf16* __restrict__ Whi,
                                               __bf16* __restrict__ Wlo) {
  size_t e = (size_t)blockIdx.x * 256 + threadIdx.x;  // (g, i, j) of W
  int j = (int)(e & (Nx - 1));
  size_t r = e >> 10;                                 // g*N + i
  int i = (int)(r & (Nx - 1));
  size_t g0 = r & ~(size_t)(Nx - 1);                  // g*N
  size_t gb = (r >> 10) * ((size_t)Nx * Nx);          // g*N*N
  float t = dis[g0 + j] * A[gb + (size_t)j * Nx + i] * dis[r];
  float m = (i == j ? 1.0f : 0.0f) - 0.95f * t;
  __bf16 h, l;
  split_bf16(m, h, l);
  Whi[e] = h;
  Wlo[e] = l;
}

// X0 = I (identity is symmetric: row-major and transposed copies identical)
__global__ __launch_bounds__(256) void init_X(float* __restrict__ Xf,
                                              __bf16* __restrict__ Xhi,
                                              __bf16* __restrict__ Xlo,
                                              __bf16* __restrict__ XThi,
                                              __bf16* __restrict__ XTlo) {
  size_t e = (size_t)blockIdx.x * 256 + threadIdx.x;
  int j = (int)(e & (Nx - 1));
  int i = (int)((e >> 10) & (Nx - 1));
  float v = (i == j) ? 1.0f : 0.0f;
  __bf16 hv = (__bf16)v, zv = (__bf16)0.0f;
  Xf[e] = v;
  Xhi[e] = hv; Xlo[e] = zv;
  XThi[e] = hv; XTlo[e] = zv;
}

// ------------------------- WMMA GEMM ---------------------------------------
// Fragment layouts per CDNA5 ISA 7.12.2 (16-bit A 16x32, B 32x16).
// A tile: row-slab [BM][BK]; B^T tile: row-slab [BN][BK] (k contiguous).

__device__ __forceinline__ v16bf load_a_frag(const __bf16* __restrict__ sA,
                                             int row0, int lane) {
  v16bf f;
  int rr = lane & 15, h8 = (lane >> 4) * 8;
  const __bf16* base = sA + (size_t)(row0 + rr) * LDT + h8;
#pragma unroll
  for (int j = 0; j < 8; ++j) {
    f[j] = base[j];          // K = h8 + 0..7
    f[j + 8] = base[16 + j]; // K = h8 + 16..23
  }
  return f;
}

__device__ __forceinline__ v16bf load_b_frag(const __bf16* __restrict__ sBT,
                                             int n0, int lane) {
  v16bf f;
  int rr = lane & 15, h8 = (lane >> 4) * 8;
  const __bf16* base = sBT + (size_t)(n0 + rr) * LDT + h8;
#pragma unroll
  for (int j = 0; j < 8; ++j) {
    f[j] = base[j];
    f[j + 8] = base[16 + j];
  }
  return f;
}

// Copy one 128x32 bf16 row-slab (global row stride Nx) into padded LDS tile.
__device__ __forceinline__ void tile_copy(const __bf16* __restrict__ g,
                                          __bf16* __restrict__ s,
                                          size_t gbase, int tid) {
#pragma unroll
  for (int ci = 0; ci < 2; ++ci) {
    int c = tid + ci * 256;          // 512 16B-chunks: row = c>>2, seg = c&3
    int row = c >> 2;
    int seg = (c & 3) * 8;
    const __bf16* gp = g + gbase + (size_t)row * Nx + seg;
    __bf16* sp = s + (size_t)row * LDT + seg;
#if HAVE_ASYNC_LDS
    __builtin_amdgcn_global_load_async_to_lds_b128((GPTR4)gp, (SPTR4)sp, 0, 0);
#else
    const uint4 v = *(const uint4*)gp;
    *(uint4*)sp = v;
#endif
  }
}

// mode 0:  C = A*B          -> write CT (hi/lo) only          (Y = W*X)
// mode 1:  C = 2*X - A*B    -> write Cf32 + C (hi/lo) + CT    (X_next)
__global__ __launch_bounds__(256)
void newton_gemm(const __bf16* __restrict__ Ahi, const __bf16* __restrict__ Alo,
                 const __bf16* __restrict__ BThi, const __bf16* __restrict__ BTlo,
                 const float* __restrict__ Xf32, float* __restrict__ Cf32,
                 __bf16* __restrict__ Chi, __bf16* __restrict__ Clo,
                 __bf16* __restrict__ CThi, __bf16* __restrict__ CTlo,
                 int mode) {
  __shared__ __bf16 sA[2][2][BM * LDT];   // [buf][hi/lo]
  __shared__ __bf16 sB[2][2][BN * LDT];

  const int tid  = threadIdx.x;
  const int lane = tid & 31;
  const int w    = tid >> 5;            // 8 waves
  const int wm   = (w & 1) * 64;
  const int wn   = (w >> 1) * 32;
  const int bn0  = blockIdx.x * BN;
  const int bm0  = blockIdx.y * BM;
  const size_t gB = (size_t)blockIdx.z * ((size_t)Nx * Nx);

  v8f acc[4][2];
#pragma unroll
  for (int a = 0; a < 4; ++a)
#pragma unroll
    for (int b = 0; b < 2; ++b) acc[a][b] = (v8f){0, 0, 0, 0, 0, 0, 0, 0};

  auto issue = [&](int buf, int k0) {
    size_t abase = gB + (size_t)bm0 * Nx + k0;
    size_t bbase = gB + (size_t)bn0 * Nx + k0;
    tile_copy(Ahi,  sA[buf][0], abase, tid);
    tile_copy(Alo,  sA[buf][1], abase, tid);
    tile_copy(BThi, sB[buf][0], bbase, tid);
    tile_copy(BTlo, sB[buf][1], bbase, tid);
  };

  issue(0, 0);                                   // prologue

  for (int kt = 0; kt < Nx / BK; ++kt) {
    const int cur = kt & 1;
    wait_async0();                               // my async stores landed
    __syncthreads();                             // everyone's landed + prev compute done
    if (kt + 1 < Nx / BK) issue(cur ^ 1, (kt + 1) * BK);

    const __bf16* cAhi = sA[cur][0];
    const __bf16* cAlo = sA[cur][1];
    const __bf16* cBhi = sB[cur][0];
    const __bf16* cBlo = sB[cur][1];

    v16bf bh[2], bl[2];
    bh[0] = load_b_frag(cBhi, wn, lane);
    bh[1] = load_b_frag(cBhi, wn + 16, lane);
    bl[0] = load_b_frag(cBlo, wn, lane);
    bl[1] = load_b_frag(cBlo, wn + 16, lane);

#pragma unroll
    for (int sm = 0; sm < 4; ++sm) {
      v16bf ah = load_a_frag(cAhi, wm + sm * 16, lane);
      v16bf al = load_a_frag(cAlo, wm + sm * 16, lane);
#pragma unroll
      for (int sn = 0; sn < 2; ++sn) {
        acc[sm][sn] = __builtin_amdgcn_wmma_f32_16x16x32_bf16(
            false, ah, false, bh[sn], (short)0, acc[sm][sn], false, false);
        acc[sm][sn] = __builtin_amdgcn_wmma_f32_16x16x32_bf16(
            false, ah, false, bl[sn], (short)0, acc[sm][sn], false, false);
        acc[sm][sn] = __builtin_amdgcn_wmma_f32_16x16x32_bf16(
            false, al, false, bh[sn], (short)0, acc[sm][sn], false, false);
      }
    }
    __syncthreads();                             // done reading cur before it refills
  }

  // epilogue: C VGPR r -> m = r + (lane>=16)*8, n = lane&15  (ISA 7.12.2)
  const int rr = lane & 15, half = lane >> 4;
#pragma unroll
  for (int sm = 0; sm < 4; ++sm)
#pragma unroll
    for (int sn = 0; sn < 2; ++sn)
#pragma unroll
      for (int r = 0; r < 8; ++r) {
        int m = bm0 + wm + sm * 16 + half * 8 + r;
        int n = bn0 + wn + sn * 16 + rr;
        size_t off  = gB + (size_t)m * Nx + n;
        size_t offT = gB + (size_t)n * Nx + m;
        float v = acc[sm][sn][r];
        __bf16 h, l;
        if (mode == 1) {
          v = 2.0f * Xf32[off] - v;
          Cf32[off] = v;
          split_bf16(v, h, l);
          Chi[off] = h;
          Clo[off] = l;
        } else {
          split_bf16(v, h, l);
        }
        CThi[offT] = h;
        CTlo[offT] = l;
      }
}

// ------------------------- top-k -------------------------------------------
// X ~ inv(M)^T: row c of X holds column c of S -> contiguous top-k.
// Repeated argmax, tie-break to smaller index (lax.top_k order).

__global__ __launch_bounds__(256)
void topk_rows(const float* __restrict__ S, int* __restrict__ outRows,
               int* __restrict__ outCols) {
  __shared__ float vals[Nx];
  __shared__ float rv[256];
  __shared__ int   ri[256];
  int b = blockIdx.x;                  // g*N + c
  int g = b >> 10;
  int c = b & (Nx - 1);
  int t = threadIdx.x;
  const float* row = S + (size_t)b * Nx;
  for (int i = t; i < Nx; i += 256) vals[i] = row[i];
  __syncthreads();
  size_t obase = (size_t)b * Kx;
  for (int k = 0; k < Kx; ++k) {
    float bv = -3.4e38f;
    int   bi = Nx;
    for (int i = t; i < Nx; i += 256) {
      float v = vals[i];
      if (v > bv || (v == bv && i < bi)) { bv = v; bi = i; }
    }
    rv[t] = bv;
    ri[t] = bi;
    __syncthreads();
    for (int s = 128; s > 0; s >>= 1) {
      if (t < s) {
        float ov = rv[t + s];
        int   oi = ri[t + s];
        if (ov > rv[t] || (ov == rv[t] && oi < ri[t])) { rv[t] = ov; ri[t] = oi; }
      }
      __syncthreads();
    }
    if (t == 0) {
      int best = ri[0];
      outRows[obase + k] = best + (g << 10);
      outCols[obase + k] = c + (g << 10);
      vals[best] = -3.4e38f;
    }
    __syncthreads();
  }
}

// ------------------------- host orchestration ------------------------------

extern "C" void kernel_launch(void* const* d_in, const int* in_sizes, int n_in,
                              void* d_out, int out_size, void* d_ws, size_t ws_size,
                              hipStream_t stream) {
  const size_t NN = (size_t)Nx * Nx;
  const size_t E  = (size_t)Gx * NN;
  const int* edge_index = (const int*)d_in[1];
  const int Etot = in_sizes[1] / 2;
  int* out = (int*)d_out;

  char* ws = (char*)d_ws;
  float*  Xf   = (float*)ws;  ws += E * sizeof(float);
  float*  Zf   = (float*)ws;  ws += E * sizeof(float);
  __bf16* Whi  = (__bf16*)ws; ws += E * 2;
  __bf16* Wlo  = (__bf16*)ws; ws += E * 2;
  __bf16* Xhi  = (__bf16*)ws; ws += E * 2;
  __bf16* Xlo  = (__bf16*)ws; ws += E * 2;
  __bf16* XThi = (__bf16*)ws; ws += E * 2;
  __bf16* XTlo = (__bf16*)ws; ws += E * 2;
  __bf16* YThi = (__bf16*)ws; ws += E * 2;
  __bf16* YTlo = (__bf16*)ws; ws += E * 2;
  __bf16* Zhi  = (__bf16*)ws; ws += E * 2;
  __bf16* Zlo  = (__bf16*)ws; ws += E * 2;
  __bf16* ZThi = (__bf16*)ws; ws += E * 2;
  __bf16* ZTlo = (__bf16*)ws; ws += E * 2;
  float*  dis  = (float*)ws;  ws += (size_t)Gx * Nx * sizeof(float);

  dim3 b256(256);
  const unsigned eb = (unsigned)(E / 256);

  zero_f32<<<eb, b256, 0, stream>>>(Xf);         // build A in Xf
  scatter_edges<<<(Etot + 255) / 256, b256, 0, stream>>>(edge_index, Etot, Xf);
  add_diag<<<Gx * Nx / 256, b256, 0, stream>>>(Xf);
  rowsum_rsqrt<<<Gx * Nx, b256, 0, stream>>>(Xf, dis);
  make_Mt<<<eb, b256, 0, stream>>>(Xf, dis, Whi, Wlo);
  init_X<<<eb, b256, 0, stream>>>(Xf, Xhi, Xlo, XThi, XTlo);

  dim3 gg(Nx / BN, Nx / BM, Gx);
  float*  xf = Xf;  __bf16* xh = Xhi;  __bf16* xl = Xlo;
  __bf16* xth = XThi; __bf16* xtl = XTlo;
  float*  zf = Zf;  __bf16* zh = Zhi;  __bf16* zl = Zlo;
  __bf16* zth = ZThi; __bf16* ztl = ZTlo;

  for (int it = 0; it < NEWTON_ITERS; ++it) {
    // Y = W * X   (only Y^T needed downstream)
    newton_gemm<<<gg, b256, 0, stream>>>(Whi, Wlo, xth, xtl,
                                         nullptr, nullptr, nullptr, nullptr,
                                         YThi, YTlo, 0);
    // Z = 2X - X*Y  (needs Z row-major, Z^T, Z f32)
    newton_gemm<<<gg, b256, 0, stream>>>(xh, xl, YThi, YTlo,
                                         xf, zf, zh, zl, zth, ztl, 1);
    float*  tf = xf;  xf = zf;  zf = tf;
    __bf16* th = xh;  xh = zh;  zh = th;
    __bf16* tl = xl;  xl = zl;  zl = tl;
    __bf16* t2 = xth; xth = zth; zth = t2;
    __bf16* t3 = xtl; xtl = ztl; ztl = t3;
  }

  topk_rows<<<Gx * Nx, b256, 0, stream>>>(xf, out, out + (size_t)Gx * Nx * Kx);
}
